// MultiHeadedAttention_30906584662141
// MI455X (gfx1250) — matrix operand order repaired and checked by reading, not compiled, not run on hardware
//
#include <hip/hip_runtime.h>
#include <stdint.h>

// ---------------------------------------------------------------------------
// Types
// ---------------------------------------------------------------------------
typedef __bf16 bf16;
typedef bf16     v16bf __attribute__((ext_vector_type(16)));
typedef float    v8f   __attribute__((ext_vector_type(8)));
typedef uint32_t u32;
typedef u32      u32x4 __attribute__((ext_vector_type(4)));
typedef float    f32x4 __attribute__((ext_vector_type(4)));

union FragBF { u32x4 u[2]; v16bf v; };

static constexpr int Bsz = 8;    // batch
static constexpr int S   = 1024; // seq len
static constexpr int D   = 1024; // model dim
static constexpr int H   = 16;   // heads
static constexpr int DK  = 64;   // head dim
static constexpr int N   = Bsz * S; // 8192 rows

// ---------------------------------------------------------------------------
// WMMA helper: D = A(16x32 bf16) x B(32x16 bf16) + C(16x16 f32)
// ---------------------------------------------------------------------------
__device__ __forceinline__ v8f wmma_bf16(v16bf a, v16bf b, v8f c) {
  return __builtin_amdgcn_wmma_f32_16x16x32_bf16(
      /*neg_a=*/false, a, /*neg_b=*/false, b,
      /*c_mod=*/(short)0, c, /*reuse_a=*/false, /*reuse_b=*/false);
}

// ---------------------------------------------------------------------------
// Fused DPP16 ROW_XMASK max butterfly: one v_max_num_f32 with a DPP modifier
// on src0 computes max(shuffle(x), x) per stage (vs mov_dpp + canonicalize +
// max when built from intrinsics). Inline asm since no builtin covers fused
// DPP arithmetic.
// ---------------------------------------------------------------------------
#define DPP_XMASK_MAX(x, m)                                                    \
  ({                                                                           \
    float _d;                                                                  \
    asm("v_max_num_f32_dpp %0, %1, %1 row_xmask:" #m                           \
        " row_mask:0xf bank_mask:0xf bound_ctrl:1"                             \
        : "=v"(_d)                                                             \
        : "v"(x));                                                             \
    _d;                                                                        \
  })

__device__ __forceinline__ float redmax16(float x) {
  x = DPP_XMASK_MAX(x, 1);
  x = DPP_XMASK_MAX(x, 2);
  x = DPP_XMASK_MAX(x, 4);
  x = DPP_XMASK_MAX(x, 8);
  return x;
}

// ---------------------------------------------------------------------------
// Fragment loaders (contiguous k = 16*(lane>>4) + s, same order for A and B)
// ---------------------------------------------------------------------------
// A-fragment (16x32, bf16 source). rowp points at this lane's row start for the
// current 32-wide K chunk; koff = 16*(lane>>4). Chunks [koff,koff+8) and [+8).
__device__ __forceinline__ v16bf load_afrag_bf16(const bf16* rowp, int koff) {
  FragBF f;
  f.u[0] = *reinterpret_cast<const u32x4*>(rowp + koff);
  f.u[1] = *reinterpret_cast<const u32x4*>(rowp + koff + 8);
  return f.v;
}

// A-fragment from f32 source, scaled then converted to bf16 in-register.
__device__ __forceinline__ v16bf load_afrag_f32(const float* rowp, int koff,
                                                float scale) {
  const f32x4* p0 = reinterpret_cast<const f32x4*>(rowp + koff);
  const f32x4* p1 = reinterpret_cast<const f32x4*>(rowp + koff + 8);
  f32x4 a = p0[0], b = p0[1], c = p1[0], d = p1[1];
  v16bf v;
#pragma unroll
  for (int i = 0; i < 4; ++i) {
    v[i]      = (bf16)(a[i] * scale);
    v[4 + i]  = (bf16)(b[i] * scale);
    v[8 + i]  = (bf16)(c[i] * scale);
    v[12 + i] = (bf16)(d[i] * scale);
  }
  return v;
}

// B-fragment (32x16 bf16): lane holds column (lane&15), K = 16*(lane>>4)..+15
// sequential -> one contiguous 32-byte chunk. colp already includes +16*(lane>>4).
__device__ __forceinline__ v16bf load_bfrag_bf16(const bf16* colp) {
  FragBF f;
  f.u[0] = *reinterpret_cast<const u32x4*>(colp);
  f.u[1] = *reinterpret_cast<const u32x4*>(colp + 8);
  return f.v;
}

// ---------------------------------------------------------------------------
// f32 -> bf16 elementwise convert (for weight matrices)
// ---------------------------------------------------------------------------
__global__ __launch_bounds__(256) void cvt_f32_bf16(const float* __restrict__ in,
                                                    bf16* __restrict__ out, int n) {
  int i = (blockIdx.x * blockDim.x + threadIdx.x) * 4;
  if (i + 3 < n) {
    f32x4 v = *reinterpret_cast<const f32x4*>(in + i);
    out[i + 0] = (bf16)v[0];
    out[i + 1] = (bf16)v[1];
    out[i + 2] = (bf16)v[2];
    out[i + 3] = (bf16)v[3];
  }
}

// ---------------------------------------------------------------------------
// GEMM: out = A(N x D) @ W^T(D x D) + bias,  W row-major [n][k].
// One wave computes a 32(M) x 64(N) output block (2 A-frags x 4 B-frags,
// 8 WMMA per K-step); K looped in steps of 32.
// MODE 0: store bf16 row-major [N,D]           (K projection)
// MODE 1: store bf16 transposed [B,H,dk,S]     (V proj -> V^T, via LDS transpose)
// MODE 2: store f32 row-major [N,D]            (output projection)
// AF32:   A operand is f32 (converted in-register) vs bf16.
// ---------------------------------------------------------------------------
template <int MODE, bool AF32>
__global__ __launch_bounds__(256) void proj_gemm(const void* __restrict__ Ap,
                                                 const bf16* __restrict__ W,
                                                 const float* __restrict__ bias,
                                                 void* __restrict__ outp) {
  const int lane = threadIdx.x & 31;
  const int lh   = lane >> 4;       // half-wave select
  const int l15  = lane & 15;
  const int koff = lh * 16;

  const int wid = (blockIdx.x * blockDim.x + threadIdx.x) >> 5;
  const int ct  = wid & 15;         // 16 column tiles of 64 (== head for MODE 1)
  const int rt  = wid >> 4;         // 256 row tiles of 32
  const int r0  = rt * 32;
  const int c0  = ct * 64;

  const float* Af = (const float*)Ap;
  const bf16*  Ab = (const bf16*)Ap;
  const int arow0 = r0 + l15;
  const int arow1 = r0 + 16 + l15;

  v8f acc[2][4] = {};

  for (int k0 = 0; k0 < D; k0 += 32) {
    v16bf a0, a1;
    if constexpr (AF32) {
      a0 = load_afrag_f32(Af + arow0 * D + k0, koff, 1.0f);
      a1 = load_afrag_f32(Af + arow1 * D + k0, koff, 1.0f);
      if (k0 + 32 < D) {
        __builtin_prefetch(Af + arow0 * D + k0 + 32, 0, 3);
        __builtin_prefetch(Af + arow1 * D + k0 + 32, 0, 3);
      }
    } else {
      a0 = load_afrag_bf16(Ab + arow0 * D + k0, koff);
      a1 = load_afrag_bf16(Ab + arow1 * D + k0, koff);
      if (k0 + 32 < D) {
        __builtin_prefetch(Ab + arow0 * D + k0 + 32, 0, 3);
        __builtin_prefetch(Ab + arow1 * D + k0 + 32, 0, 3);
      }
    }
#pragma unroll
    for (int t = 0; t < 4; ++t) {
      const int n = c0 + t * 16 + l15;
      v16bf b = load_bfrag_bf16(W + n * D + k0 + lh * 16);
      acc[0][t] = wmma_bf16(a0, b, acc[0][t]);
      acc[1][t] = wmma_bf16(a1, b, acc[1][t]);
    }
  }

  bf16*  outbf = (bf16*)outp;
  float* outf  = (float*)outp;

  if constexpr (MODE == 1) {
    // Transpose the 32(s) x 64(d) wave tile through wave-private LDS so the
    // global stores run contiguously along s (b128 chunks). Padded stride 40
    // spreads LDS banks; wave-local region -> no barrier (DS in-order/wave).
    __shared__ __attribute__((aligned(16))) bf16 tl[8 * 64 * 40];
    bf16* tw = tl + (threadIdx.x >> 5) * (64 * 40);
#pragma unroll
    for (int half = 0; half < 2; ++half) {
#pragma unroll
      for (int t = 0; t < 4; ++t) {
        const int ddl = t * 16 + l15;          // 0..63 within head
        const float bi = bias[c0 + ddl];
#pragma unroll
        for (int r = 0; r < 8; ++r) {
          const int ssl = half * 16 + r + 8 * lh;  // 0..31 within tile
          tw[ddl * 40 + ssl] = (bf16)(acc[half][t][r] + bi);
        }
      }
    }
    const int bb = r0 >> 10;            // batch (row tiles never straddle)
    const int sb = r0 & (S - 1);        // s base
    bf16* ob = outbf + (bb * H + ct) * DK * S + sb;
#pragma unroll
    for (int i = 0; i < 8; ++i) {       // 256 16B-chunks / 32 lanes
      const int c  = i * 32 + lane;
      const int dd = c >> 2;            // 0..63
      const int cc = (c & 3) * 8;       // s sub-offset 0/8/16/24
      u32x4 vv = *reinterpret_cast<const u32x4*>(tw + dd * 40 + cc);
      *reinterpret_cast<u32x4*>(ob + dd * S + cc) = vv;
    }
  } else {
#pragma unroll
    for (int half = 0; half < 2; ++half) {
#pragma unroll
      for (int t = 0; t < 4; ++t) {
        const int col = c0 + t * 16 + l15;
        const float bi = bias[col];
#pragma unroll
        for (int r = 0; r < 8; ++r) {
          const int row = r0 + half * 16 + r + 8 * lh;
          const float val = acc[half][t][r] + bi;
          if constexpr (MODE == 0) outbf[row * D + col] = (bf16)val;
          else                     outf[row * D + col] = val;
        }
      }
    }
  }
}

// ---------------------------------------------------------------------------
// Flash-style attention. One wave owns 16 q-rows of one (b,h); loops over S in
// 32-wide key chunks with online softmax; PV accumulated via WMMA with P
// re-layouted (C-layout -> A-layout) through per-wave LDS. The softmax
// denominator is accumulated with an extra WMMA against a ones-matrix
// (row-sums land in every column of sacc), so only the max tree runs on DPP.
// q: f32 [B,S,D] (raw query; 1/sqrt(dk) folded into the bf16 conversion)
// kbf: bf16 [B,S,D] = key@Wk^T+bk
// vt:  bf16 [B,H,DK,S] = (value@Wv^T+bv) transposed per head
// xbf: bf16 [B,S,D] output, consumed by out-projection
// ---------------------------------------------------------------------------
__global__ __launch_bounds__(128) void attn_kernel(const float* __restrict__ q,
                                                   const bf16* __restrict__ kbf,
                                                   const bf16* __restrict__ vt,
                                                   bf16* __restrict__ xbf) {
  __shared__ __attribute__((aligned(16))) bf16 plds[4 * 16 * 32];

  const int lane = threadIdx.x & 31;
  const int wb   = threadIdx.x >> 5;   // wave in block (4)
  const int lh   = lane >> 4;
  const int l15  = lane & 15;
  const int koff = lh * 16;

  const int wid = (blockIdx.x * blockDim.x + threadIdx.x) >> 5;
  const int qt  = wid & 63;            // 64 q-tiles per (b,h)
  const int bh  = wid >> 6;            // 0..127
  const int b   = bh >> 4;
  const int h   = bh & 15;

  const int rowbase = b * S + qt * 16; // flat q-row base (b*S + s)
  const int kcol    = h * DK;          // head column offset in [N,D] layout

  // Preload q A-fragments for the two 32-wide chunks of dk=64; scale by 1/8.
  const float* qrow = q + (rowbase + l15) * D + kcol;
  const v16bf qa0 = load_afrag_f32(qrow + 0, koff, 0.125f);
  const v16bf qa1 = load_afrag_f32(qrow + 32, koff, 0.125f);

  // Ones B-matrix for row-sum-via-WMMA.
  v16bf ones;
#pragma unroll
  for (int i = 0; i < 16; ++i) ones[i] = (bf16)1.0f;

  float m[8];
  v8f xacc[4] = {};
  v8f sacc = {};                        // row-sum accumulator (all cols equal)
#pragma unroll
  for (int r = 0; r < 8; ++r) m[r] = -1e30f;

  bf16* lp = plds + wb * 512;
  const bf16* vth = vt + bh * DK * S;
  const bf16* kbb = kbf + b * S * D;

  for (int j0 = 0; j0 < S; j0 += 32) {
    // ---- scores: 16x32 block of (q/8).k^T -----------------------------
    const bf16* kr0 = kbb + (j0 + l15) * D + kcol;        // cols j0..j0+15
    const bf16* kr1 = kbb + (j0 + 16 + l15) * D + kcol;   // cols j0+16..j0+31
    if (j0 + 32 < S) {  // uniform branch; prefetch next chunk
      __builtin_prefetch(kbb + (j0 + 32 + l15) * D + kcol, 0, 3);
      __builtin_prefetch(vth + l15 * S + j0 + 32, 0, 3);
    }
    v8f slo = {}, shi = {};
    slo = wmma_bf16(qa0, load_bfrag_bf16(kr0 + 0  + lh * 16), slo);
    shi = wmma_bf16(qa0, load_bfrag_bf16(kr1 + 0  + lh * 16), shi);
    slo = wmma_bf16(qa1, load_bfrag_bf16(kr0 + 32 + lh * 16), slo);
    shi = wmma_bf16(qa1, load_bfrag_bf16(kr1 + 32 + lh * 16), shi);

    // ---- online softmax update (rows live across 16-lane groups) ------
#pragma unroll
    for (int r = 0; r < 8; ++r) {
      const float sl = slo[r];
      const float sh = shi[r];
      const float cmax  = redmax16(fmaxf(sl, sh));
      const float mn    = fmaxf(m[r], cmax);
      const float alpha = __expf(m[r] - mn);
      m[r] = mn;
      slo[r] = __expf(sl - mn);
      shi[r] = __expf(sh - mn);
      sacc[r] *= alpha;
#pragma unroll
      for (int t = 0; t < 4; ++t) xacc[t][r] *= alpha;
    }

    // ---- reshape P: C-layout -> A-layout via per-wave LDS -------------
#pragma unroll
    for (int r = 0; r < 8; ++r) {
      const int row = r + 8 * lh;
      lp[row * 32 + l15]      = (bf16)slo[r];
      lp[row * 32 + 16 + l15] = (bf16)shi[r];
    }
    const v16bf pa = load_afrag_bf16(lp + l15 * 32, koff);  // same-wave DS: in order

    // ---- PV + row-sums: contraction over this j chunk -----------------
#pragma unroll
    for (int t = 0; t < 4; ++t) {
      const bf16* vr = vth + (t * 16 + l15) * S + j0 + lh * 16;
      xacc[t] = wmma_bf16(pa, load_bfrag_bf16(vr), xacc[t]);
    }
    sacc = wmma_bf16(pa, ones, sacc);   // row-sum of P into every column
  }

  // ---- normalize and store bf16 [N,D] -----------------------------------
#pragma unroll
  for (int r = 0; r < 8; ++r) {
    const float inv = 1.0f / sacc[r];
    const int row = rowbase + r + 8 * lh;
#pragma unroll
    for (int t = 0; t < 4; ++t)
      xbf[row * D + kcol + t * 16 + l15] = (bf16)(xacc[t][r] * inv);
  }
}

// ---------------------------------------------------------------------------
// Host-side launch
// ---------------------------------------------------------------------------
extern "C" void kernel_launch(void* const* d_in, const int* in_sizes, int n_in,
                              void* d_out, int out_size, void* d_ws, size_t ws_size,
                              hipStream_t stream) {
  (void)in_sizes; (void)n_in; (void)out_size; (void)ws_size;

  const float* query = (const float*)d_in[0];
  const float* key   = (const float*)d_in[1];
  const float* value = (const float*)d_in[2];
  const float* Wk    = (const float*)d_in[3];
  const float* bk    = (const float*)d_in[4];
  const float* Wv    = (const float*)d_in[5];
  const float* bv    = (const float*)d_in[6];
  const float* Wo    = (const float*)d_in[7];
  const float* bo    = (const float*)d_in[8];
  float* out = (float*)d_out;

  // Workspace carve-up (54 MiB total)
  char* ws = (char*)d_ws;
  bf16* wkbf = (bf16*)(ws + (size_t(0)  << 20));  // 2 MiB  (1M bf16)
  bf16* wvbf = (bf16*)(ws + (size_t(2)  << 20));  // 2 MiB
  bf16* wobf = (bf16*)(ws + (size_t(4)  << 20));  // 2 MiB
  bf16* kbf  = (bf16*)(ws + (size_t(6)  << 20));  // 16 MiB (8M bf16)
  bf16* vtbf = (bf16*)(ws + (size_t(22) << 20));  // 16 MiB
  bf16* xbf  = (bf16*)(ws + (size_t(38) << 20));  // 16 MiB

  const int WEL = D * D;  // 1M elements per weight matrix

  // 1) weights -> bf16
  cvt_f32_bf16<<<WEL / 4 / 256, 256, 0, stream>>>(Wk, wkbf, WEL);
  cvt_f32_bf16<<<WEL / 4 / 256, 256, 0, stream>>>(Wv, wvbf, WEL);
  cvt_f32_bf16<<<WEL / 4 / 256, 256, 0, stream>>>(Wo, wobf, WEL);

  // 2) K = key @ Wk^T + bk   (bf16, row-major [N,D])
  //    V^T = (value @ Wv^T + bv) transposed per head (bf16, [B,H,DK,S])
  // grid: (N/32)*(D/64) = 4096 wave tiles, 8 waves / 256-thread block
  proj_gemm<0, true ><<<512, 256, 0, stream>>>(key,   wkbf, bk, kbf);
  proj_gemm<1, true ><<<512, 256, 0, stream>>>(value, wvbf, bv, vtbf);

  // 3) attention: B*H*(S/16) = 8192 waves, 4 waves / 128-thread block
  attn_kernel<<<2048, 128, 0, stream>>>(query, kbf, vtbf, xbf);

  // 4) out = X @ Wo^T + bo (f32)
  proj_gemm<2, false><<<512, 256, 0, stream>>>(xbf, wobf, bo, out);
}
